// PoolNet_61607010894040
// MI455X (gfx1250) — matrix operand order repaired and checked by reading, not compiled, not run on hardware
//
#include <hip/hip_runtime.h>
#include <math.h>

// ---------------- problem constants ----------------
static constexpr int BG   = 8;
static constexpr int NPG0 = 8192;
static constexpr int DEG  = 16;
static constexpr int NE   = BG * NPG0 * DEG;   // 1,048,576 edges
static constexpr int KP1  = 2458;
static constexpr int KP2  = 738;
static constexpr int NT1  = BG * NPG0;         // 65536 nodes
static constexpr int NT2  = BG * KP1;          // 19664
static constexpr int NT3  = BG * KP2;          // 5904
static constexpr int NCLS = 10;

typedef __attribute__((ext_vector_type(16))) _Float16 v16h;
typedef __attribute__((ext_vector_type(8)))  float    v8f;

// ---------------- device helpers ----------------
__device__ __forceinline__ unsigned f2o(float f) {
  unsigned u = __float_as_uint(f);
  return (u & 0x80000000u) ? ~u : (u | 0x80000000u);
}
__device__ __forceinline__ float o2f(unsigned e) {
  unsigned u = (e & 0x80000000u) ? (e & 0x7FFFFFFFu) : ~e;
  return __uint_as_float(u);
}
__device__ __forceinline__ float elu1(float v) { return v > 0.f ? v : (__expf(v) - 1.f); }

// ---------------- WMMA GEMM: C[M,N] = act(A[M,K] @ B[K,N] + bias) ----------------
// fp32 inputs converted to f16 fragments, f32 accumulate via v_wmma_f32_16x16x32_f16.
// REQUIREMENTS: M%16==0, N%16==0, K%32==0, A rows 16-byte aligned (A base 256B-aligned
// and K*4 a multiple of 32B). No per-element guards -> loads form clean clauses.
// Bias presence is compile-time so the epilogue is branch-free (bias loaded once;
// col = tn*16 + r is invariant across the 8 output rows of this lane).
// One wave32 per 16x16 output tile; blockDim must be a multiple of 32.
template <int ACT, bool HAS_BIAS>
__global__ void gemm_wmma_f16(const float* __restrict__ A, const float* __restrict__ Bm,
                              const float* __restrict__ bias, float* __restrict__ C,
                              int M, int N, int K) {
  int gtid = blockIdx.x * blockDim.x + threadIdx.x;
  int wave = gtid >> 5;
  int lane = gtid & 31;
  int tilesN = N >> 4;
  int tilesM = M >> 4;
  if (wave >= tilesM * tilesN) return;   // uniform per wave
  int tm = wave / tilesN;
  int tn = wave % tilesN;
  int r  = lane & 15;        // row-in-tile (A) / col-in-tile (B,C)
  int hi = lane >> 4;        // lane-half selector
  const int khalfA = hi * 8;
  v8f acc = {};
  const float* Arow = A + (size_t)(tm * 16 + r) * K;
  const float* Bcol = Bm + (size_t)(tn * 16 + r);
  for (int kb = 0; kb < K; kb += 32) {
    // A 16x32 f16 layout: lanes 0-15 row=r hold K {0..7,16..23}; lanes 16-31 {8..15,24..31}
    const float4 a0 = *(const float4*)(Arow + kb + khalfA);
    const float4 a1 = *(const float4*)(Arow + kb + khalfA + 4);
    const float4 a2 = *(const float4*)(Arow + kb + 16 + khalfA);
    const float4 a3 = *(const float4*)(Arow + kb + 16 + khalfA + 4);
    v16h a;
    a[0]  = (_Float16)a0.x; a[1]  = (_Float16)a0.y; a[2]  = (_Float16)a0.z; a[3]  = (_Float16)a0.w;
    a[4]  = (_Float16)a1.x; a[5]  = (_Float16)a1.y; a[6]  = (_Float16)a1.z; a[7]  = (_Float16)a1.w;
    a[8]  = (_Float16)a2.x; a[9]  = (_Float16)a2.y; a[10] = (_Float16)a2.z; a[11] = (_Float16)a2.w;
    a[12] = (_Float16)a3.x; a[13] = (_Float16)a3.y; a[14] = (_Float16)a3.z; a[15] = (_Float16)a3.w;
    // B 32x16 f16 layout: lanes 0-15 K {0..7,16..23}; lanes 16-31 {8..15,24..31}, col = lane%16
    v16h b;
#pragma unroll
    for (int i = 0; i < 16; ++i) {
      b[i] = (_Float16)Bcol[(size_t)(kb + hi * 8 + i + (i & 8)) * N];
    }
    acc = __builtin_amdgcn_wmma_f32_16x16x32_f16(false, a, false, b, (short)0, acc,
                                                 false, false);
  }
  // C/D layout: VGPR i -> row = i + 8*(lane/16), col = lane%16
  int col = tn * 16 + r;
  float bv = 0.f;
  if constexpr (HAS_BIAS) bv = bias[col];
  float* Cbase = C + (size_t)(tm * 16 + 8 * hi) * N + col;
#pragma unroll
  for (int i = 0; i < 8; ++i) {
    float v = acc[i] + bv;
    if (ACT == 1) v = elu1(v);
    Cbase[(size_t)i * N] = v;
  }
}

// ---------------- fills ----------------
__global__ void fill_f32(float* p, float v, int n) {
  int i = blockIdx.x * blockDim.x + threadIdx.x;
  if (i < n) p[i] = v;
}
__global__ void fill_i32(int* p, int v, int n) {
  int i = blockIdx.x * blockDim.x + threadIdx.x;
  if (i < n) p[i] = v;
}
__global__ void fill_enc(unsigned* p, float v, int n) {
  int i = blockIdx.x * blockDim.x + threadIdx.x;
  if (i < n) p[i] = f2o(v);
}

// ---------------- GAT pieces ----------------
// al[i,h] = sum_f xp[i, h*F+f] * a[h,f]
__global__ void gat_alpha(const float* __restrict__ xp, const float* __restrict__ a,
                          float* __restrict__ al, int n, int H, int F) {
  int idx = blockIdx.x * blockDim.x + threadIdx.x;
  if (idx >= n * H) return;
  int i = idx / H, h = idx % H;
  const float* xr = xp + (size_t)i * H * F + h * F;
  const float* ar = a + h * F;
  float s = 0.f;
  for (int f = 0; f < F; ++f) s += xr[f] * ar[f];
  al[idx] = s;
}

// segment-max of leaky_relu(al_s[src]+al_d[dst]) into emax[dst] (ordered-uint atomicMax).
// idx < E_: real edge; idx >= E_: self loop for node idx-E_.
__global__ void gat_edge_max(const int* __restrict__ src, const int* __restrict__ dst,
                             const int* __restrict__ mask, const float* __restrict__ al_s,
                             const float* __restrict__ al_d, unsigned* __restrict__ emax,
                             int E_, int n, int H) {
  int idx = blockIdx.x * blockDim.x + threadIdx.x;
  if (idx >= E_ + n) return;
  int s, d, m;
  if (idx < E_) { s = src[idx]; d = dst[idx]; m = mask ? mask[idx] : 1; }
  else          { s = d = idx - E_; m = 1; }
  for (int h = 0; h < H; ++h) {
    float e = al_s[s * H + h] + al_d[d * H + h];
    e = e > 0.f ? e : 0.2f * e;
    if (!m) e = -1e9f;  // masked edges still feed -1e9 into segment_max (matches reference)
    atomicMax(&emax[d * H + h], f2o(e));
  }
}

// ex = exp(e - emax[d]); den[d] += ex; out[d,:] += ex * xp[s,:]
__global__ void gat_edge_msg(const int* __restrict__ src, const int* __restrict__ dst,
                             const int* __restrict__ mask, const float* __restrict__ al_s,
                             const float* __restrict__ al_d, const unsigned* __restrict__ emax,
                             float* __restrict__ den, const float* __restrict__ xp,
                             float* __restrict__ out, int E_, int n, int H, int F) {
  int idx = blockIdx.x * blockDim.x + threadIdx.x;
  if (idx >= E_ + n) return;
  int s, d, m;
  if (idx < E_) { s = src[idx]; d = dst[idx]; m = mask ? mask[idx] : 1; }
  else          { s = d = idx - E_; m = 1; }
  if (!m) return;  // masked edges contribute 0
  for (int h = 0; h < H; ++h) {
    float e = al_s[s * H + h] + al_d[d * H + h];
    e = e > 0.f ? e : 0.2f * e;
    float ex = __expf(e - o2f(emax[d * H + h]));
    atomicAdd(&den[d * H + h], ex);
    const float* xs = xp + (size_t)s * H * F + h * F;
    float* od = out + (size_t)d * H * F + h * F;
    __builtin_prefetch(xs, 0, 1);
    for (int f = 0; f < F; ++f) atomicAdd(&od[f], ex * xs[f]);
  }
}

__global__ void gat_finalize(float* __restrict__ out, const float* __restrict__ den,
                             const float* __restrict__ bias, int n, int H, int F) {
  int idx = blockIdx.x * blockDim.x + threadIdx.x;
  if (idx >= n * H * F) return;
  int i = idx / (H * F);
  int rem = idx % (H * F);
  int h = rem / F;
  out[idx] = out[idx] / den[i * H + h] + bias[rem];
}

// ---------------- InstanceNorm + ELU ----------------
__global__ void inorm_stats(const float* __restrict__ x, int npg, int C,
                            float* __restrict__ mu, float* __restrict__ rstd) {
  int g = blockIdx.x / C, c = blockIdx.x % C;
  const float* xb = x + (size_t)g * npg * C + c;
  float s = 0.f, q = 0.f;
  for (int i = threadIdx.x; i < npg; i += blockDim.x) {
    float v = xb[(size_t)i * C];
    s += v; q += v * v;
  }
  __shared__ float rs[256], rq[256];
  rs[threadIdx.x] = s; rq[threadIdx.x] = q;
  __syncthreads();
  for (int t = blockDim.x >> 1; t > 0; t >>= 1) {
    if ((int)threadIdx.x < t) { rs[threadIdx.x] += rs[threadIdx.x + t]; rq[threadIdx.x] += rq[threadIdx.x + t]; }
    __syncthreads();
  }
  if (threadIdx.x == 0) {
    float m = rs[0] / (float)npg;
    float var = rq[0] / (float)npg - m * m;
    var = var > 0.f ? var : 0.f;
    mu[blockIdx.x] = m;
    rstd[blockIdx.x] = rsqrtf(var + 1e-5f);
  }
}
__global__ void inorm_apply_elu(float* __restrict__ x, const float* __restrict__ mu,
                                const float* __restrict__ rstd, int npg, int C, int total) {
  int idx = blockIdx.x * blockDim.x + threadIdx.x;
  if (idx >= total) return;
  int i = idx / C, c = idx % C, g = i / npg;
  float v = (x[idx] - mu[g * C + c]) * rstd[g * C + c];
  x[idx] = elu1(v);
}

// ---------------- TopK pooling ----------------
__global__ void topk_score(const float* __restrict__ x, const float* __restrict__ p,
                           float* __restrict__ score, int n, int F) {
  int i = blockIdx.x * blockDim.x + threadIdx.x;
  if (i >= n) return;
  float pn = 0.f;
  for (int f = 0; f < F; ++f) pn += p[f] * p[f];
  pn = sqrtf(pn);
  const float* xr = x + (size_t)i * F;
  float s = 0.f;
  for (int f = 0; f < F; ++f) s += xr[f] * p[f];
  score[i] = tanhf(s / pn);
}

// Per-graph radix select of k-th largest score (encoded); also counts strictly-greater.
__global__ void radix_select(const float* __restrict__ score, int npg, int k,
                             unsigned* __restrict__ Tenc, int* __restrict__ cntgt) {
  int g = blockIdx.x;
  const float* s = score + (size_t)g * npg;
  __shared__ unsigned hist[256];
  __shared__ unsigned sh_prefix;
  __shared__ int sh_remain;
  __shared__ int sh_cg;
  if (threadIdx.x == 0) { sh_prefix = 0u; sh_remain = k; sh_cg = 0; }
  __syncthreads();
  for (int d = 3; d >= 0; --d) {
    int shift = d * 8;
    unsigned himask = (d == 3) ? 0u : (0xFFFFFFFFu << ((d + 1) * 8));
    hist[threadIdx.x] = 0u;
    __syncthreads();
    unsigned pfx = sh_prefix;
    for (int i = threadIdx.x; i < npg; i += blockDim.x) {
      unsigned u = f2o(s[i]);
      if ((u & himask) == (pfx & himask)) atomicAdd(&hist[(u >> shift) & 255u], 1u);
    }
    __syncthreads();
    if (threadIdx.x == 0) {
      int rem = sh_remain;
      int b = 255;
      for (; b > 0; --b) {
        int c = (int)hist[b];
        if (c >= rem) break;
        rem -= c;
      }
      sh_prefix |= ((unsigned)b) << shift;
      sh_remain = rem;
    }
    __syncthreads();
  }
  unsigned T = sh_prefix;
  int local = 0;
  for (int i = threadIdx.x; i < npg; i += blockDim.x)
    if (f2o(s[i]) > T) ++local;
  atomicAdd(&sh_cg, local);
  __syncthreads();
  if (threadIdx.x == 0) { Tenc[g] = T; cntgt[g] = sh_cg; }
}

// Select top-k nodes, write newid map (-1 if dropped) and xnew = x * score.
__global__ void topk_compact(const float* __restrict__ x, const float* __restrict__ score,
                             int npg, int k, int C, const unsigned* __restrict__ Tenc,
                             const int* __restrict__ cntgt, int* __restrict__ ctrA,
                             int* __restrict__ ctrB, int* __restrict__ newid,
                             float* __restrict__ xnew, int total) {
  int i = blockIdx.x * blockDim.x + threadIdx.x;
  if (i >= total) return;
  int g = i / npg;
  unsigned u = f2o(score[i]);
  unsigned T = Tenc[g];
  int slot = -1;
  if (u > T) {
    slot = atomicAdd(&ctrA[g], 1);
  } else if (u == T) {
    int e = atomicAdd(&ctrB[g], 1);
    int need = k - cntgt[g];
    if (e < need) slot = cntgt[g] + e;
  }
  if (slot >= 0) {
    int ni = g * k + slot;
    newid[i] = ni;
    float sv = score[i];
    const float* xr = x + (size_t)i * C;
    float* xo = xnew + (size_t)ni * C;
    for (int c = 0; c < C; ++c) xo[c] = xr[c] * sv;
  } else {
    newid[i] = -1;
  }
}

__global__ void remap_edges(const int* __restrict__ src, const int* __restrict__ dst,
                            const int* __restrict__ maskIn, const int* __restrict__ newid,
                            int* __restrict__ srcOut, int* __restrict__ dstOut,
                            int* __restrict__ maskOut, int E_) {
  int e = blockIdx.x * blockDim.x + threadIdx.x;
  if (e >= E_) return;
  int m = maskIn ? maskIn[e] : 1;
  int ns = newid[src[e]];
  int nd = newid[dst[e]];
  maskOut[e] = (m && ns >= 0 && nd >= 0) ? 1 : 0;
  srcOut[e] = ns < 0 ? 0 : ns;
  dstOut[e] = nd < 0 ? 0 : nd;
}

// ---------------- Global attention readout ----------------
__global__ void att_readout(const float* __restrict__ gate, const float* __restrict__ h,
                            float* __restrict__ gout, int npg, int C) {
  int g = blockIdx.x;
  const float* gt = gate + (size_t)g * npg;
  __shared__ float red[256];
  float mx = -1e30f;
  for (int i = threadIdx.x; i < npg; i += blockDim.x) mx = fmaxf(mx, gt[i]);
  red[threadIdx.x] = mx;
  __syncthreads();
  for (int t = blockDim.x >> 1; t > 0; t >>= 1) {
    if ((int)threadIdx.x < t) red[threadIdx.x] = fmaxf(red[threadIdx.x], red[threadIdx.x + t]);
    __syncthreads();
  }
  mx = red[0];
  __syncthreads();
  float se = 0.f;
  for (int i = threadIdx.x; i < npg; i += blockDim.x) se += __expf(gt[i] - mx);
  red[threadIdx.x] = se;
  __syncthreads();
  for (int t = blockDim.x >> 1; t > 0; t >>= 1) {
    if ((int)threadIdx.x < t) red[threadIdx.x] += red[threadIdx.x + t];
    __syncthreads();
  }
  se = red[0];
  __syncthreads();
  for (int c = threadIdx.x; c < C; c += blockDim.x) {
    float acc = 0.f;
    for (int i = 0; i < npg; ++i) acc += __expf(gt[i] - mx) * h[((size_t)g * npg + i) * C + c];
    gout[g * C + c] = acc / se;
  }
}

// ---------------- small dense (N or K not WMMA-friendly) ----------------
__global__ void dense_scalar(const float* __restrict__ A, const float* __restrict__ W,
                             const float* __restrict__ bias, float* __restrict__ C,
                             int M, int N, int K, int act) {
  int idx = blockIdx.x * blockDim.x + threadIdx.x;
  if (idx >= M * N) return;
  int m = idx / N, nn = idx % N;
  float acc = bias ? bias[nn] : 0.f;
  const float* ar = A + (size_t)m * K;
  for (int k = 0; k < K; ++k) acc += ar[k] * W[(size_t)k * N + nn];
  if (act) acc = elu1(acc);
  C[idx] = acc;
}

__global__ void log_softmax_rows(const float* __restrict__ z, float* __restrict__ out,
                                 int rows, int cols) {
  int r = blockIdx.x * blockDim.x + threadIdx.x;
  if (r >= rows) return;
  float mx = -1e30f;
  for (int c = 0; c < cols; ++c) mx = fmaxf(mx, z[r * cols + c]);
  float se = 0.f;
  for (int c = 0; c < cols; ++c) se += __expf(z[r * cols + c] - mx);
  float l = logf(se);
  for (int c = 0; c < cols; ++c) out[r * cols + c] = z[r * cols + c] - mx - l;
}

// ---------------- host ----------------
static inline int cdiv(int a, int b) { return (a + b - 1) / b; }

static void launch_gemm(int act, const float* A, const float* B, const float* bias,
                        float* C, int M, int N, int K, hipStream_t stream) {
  // K must be a multiple of 32 here (WMMA path, no guards)
  int tiles = (M / 16) * (N / 16);
  int threads = tiles * 32;
  int blocks = cdiv(threads, 128);
  if (act) {
    if (bias) gemm_wmma_f16<1, true ><<<blocks, 128, 0, stream>>>(A, B, bias, C, M, N, K);
    else      gemm_wmma_f16<1, false><<<blocks, 128, 0, stream>>>(A, B, bias, C, M, N, K);
  } else {
    if (bias) gemm_wmma_f16<0, true ><<<blocks, 128, 0, stream>>>(A, B, bias, C, M, N, K);
    else      gemm_wmma_f16<0, false><<<blocks, 128, 0, stream>>>(A, B, bias, C, M, N, K);
  }
}

extern "C" void kernel_launch(void* const* d_in, const int* in_sizes, int n_in,
                              void* d_out, int out_size, void* d_ws, size_t ws_size,
                              hipStream_t stream) {
  (void)in_sizes; (void)n_in; (void)out_size; (void)ws_size;

  const float* pos  = (const float*)d_in[0];
  const int*   src  = (const int*)d_in[1];
  const int*   dst  = (const int*)d_in[2];
  const float* W1   = (const float*)d_in[3];
  const float* a_s1 = (const float*)d_in[4];
  const float* a_d1 = (const float*)d_in[5];
  const float* b1   = (const float*)d_in[6];
  const float* p1   = (const float*)d_in[7];
  const float* W2   = (const float*)d_in[8];
  const float* a_s2 = (const float*)d_in[9];
  const float* a_d2 = (const float*)d_in[10];
  const float* b2   = (const float*)d_in[11];
  const float* p2   = (const float*)d_in[12];
  const float* W3   = (const float*)d_in[13];
  const float* a_s3 = (const float*)d_in[14];
  const float* a_d3 = (const float*)d_in[15];
  const float* b3   = (const float*)d_in[16];
  const float* gw1  = (const float*)d_in[17];
  const float* gb1  = (const float*)d_in[18];
  const float* gw2  = (const float*)d_in[19];
  const float* gb2  = (const float*)d_in[20];
  const float* aw   = (const float*)d_in[21];
  const float* ab   = (const float*)d_in[22];
  const float* w1   = (const float*)d_in[23];
  const float* bb1  = (const float*)d_in[24];
  const float* w2   = (const float*)d_in[25];
  const float* bb2  = (const float*)d_in[26];
  const float* w3   = (const float*)d_in[27];
  const float* bb3  = (const float*)d_in[28];

  // bump allocator over d_ws (all buffers 256B-aligned)
  char* base = (char*)d_ws;
  size_t off = 0;
  auto alloc = [&](size_t bytes) -> void* {
    void* p = base + off;
    off += (bytes + 255) & ~(size_t)255;
    return p;
  };

  float*    xp1    = (float*)alloc((size_t)NT1 * 32 * 4);
  float*    alS1   = (float*)alloc((size_t)NT1 * 2 * 4);
  float*    alD1   = (float*)alloc((size_t)NT1 * 2 * 4);
  unsigned* emax1  = (unsigned*)alloc((size_t)NT1 * 2 * 4);
  float*    den1   = (float*)alloc((size_t)NT1 * 2 * 4);
  float*    gat1   = (float*)alloc((size_t)NT1 * 32 * 4);
  float*    score1 = (float*)alloc((size_t)NT1 * 4);
  int*      newid1 = (int*)alloc((size_t)NT1 * 4);
  float*    x2     = (float*)alloc((size_t)NT2 * 32 * 4);
  int*      src2   = (int*)alloc((size_t)NE * 4);
  int*      dst2   = (int*)alloc((size_t)NE * 4);
  int*      mask2  = (int*)alloc((size_t)NE * 4);
  float*    xp2    = (float*)alloc((size_t)NT2 * 128 * 4);
  float*    alS2   = (float*)alloc((size_t)NT2 * 2 * 4);
  float*    alD2   = (float*)alloc((size_t)NT2 * 2 * 4);
  unsigned* emax2  = (unsigned*)alloc((size_t)NT2 * 2 * 4);
  float*    den2   = (float*)alloc((size_t)NT2 * 2 * 4);
  float*    gat2   = (float*)alloc((size_t)NT2 * 128 * 4);
  float*    score2 = (float*)alloc((size_t)NT2 * 4);
  int*      newid2 = (int*)alloc((size_t)NT2 * 4);
  float*    x3     = (float*)alloc((size_t)NT3 * 128 * 4);
  int*      src3   = (int*)alloc((size_t)NE * 4);
  int*      dst3   = (int*)alloc((size_t)NE * 4);
  int*      mask3  = (int*)alloc((size_t)NE * 4);
  float*    xp3    = (float*)alloc((size_t)NT3 * 256 * 4);
  float*    alS3   = (float*)alloc((size_t)NT3 * 4);
  float*    alD3   = (float*)alloc((size_t)NT3 * 4);
  unsigned* emax3  = (unsigned*)alloc((size_t)NT3 * 4);
  float*    den3   = (float*)alloc((size_t)NT3 * 4);
  float*    gat3   = (float*)alloc((size_t)NT3 * 256 * 4);
  float*    mu     = (float*)alloc((size_t)BG * 256 * 4);
  float*    rstd   = (float*)alloc((size_t)BG * 256 * 4);
  unsigned* Tenc   = (unsigned*)alloc(BG * 4);
  int*      cntgt  = (int*)alloc(BG * 4);
  int*      ctrA   = (int*)alloc(BG * 4);
  int*      ctrB   = (int*)alloc(BG * 4);
  float*    gate1  = (float*)alloc((size_t)NT3 * 64 * 4);
  float*    gateS  = (float*)alloc((size_t)NT3 * 4);
  float*    hbuf   = (float*)alloc((size_t)NT3 * 256 * 4);
  float*    gvec   = (float*)alloc((size_t)BG * 256 * 4);
  float*    z1     = (float*)alloc((size_t)BG * 512 * 4);
  float*    z2     = (float*)alloc((size_t)BG * 128 * 4);
  float*    logits = (float*)alloc((size_t)BG * NCLS * 4);

  const int TB = 256;

  // ================= Layer 1: GAT(3 -> 32, H=2, F=16) =================
  // K=3 is not WMMA-friendly (and trivial work): scalar dense.
  dense_scalar<<<cdiv(NT1 * 32, TB), TB, 0, stream>>>(pos, W1, nullptr, xp1, NT1, 32, 3, 0);
  gat_alpha<<<cdiv(NT1 * 2, TB), TB, 0, stream>>>(xp1, a_s1, alS1, NT1, 2, 16);
  gat_alpha<<<cdiv(NT1 * 2, TB), TB, 0, stream>>>(xp1, a_d1, alD1, NT1, 2, 16);
  fill_enc<<<cdiv(NT1 * 2, TB), TB, 0, stream>>>(emax1, -1e30f, NT1 * 2);
  fill_f32<<<cdiv(NT1 * 2, TB), TB, 0, stream>>>(den1, 0.f, NT1 * 2);
  fill_f32<<<cdiv(NT1 * 32, TB), TB, 0, stream>>>(gat1, 0.f, NT1 * 32);
  gat_edge_max<<<cdiv(NE + NT1, TB), TB, 0, stream>>>(src, dst, nullptr, alS1, alD1, emax1,
                                                      NE, NT1, 2);
  gat_edge_msg<<<cdiv(NE + NT1, TB), TB, 0, stream>>>(src, dst, nullptr, alS1, alD1, emax1,
                                                      den1, xp1, gat1, NE, NT1, 2, 16);
  gat_finalize<<<cdiv(NT1 * 32, TB), TB, 0, stream>>>(gat1, den1, b1, NT1, 2, 16);
  inorm_stats<<<BG * 32, TB, 0, stream>>>(gat1, NPG0, 32, mu, rstd);
  inorm_apply_elu<<<cdiv(NT1 * 32, TB), TB, 0, stream>>>(gat1, mu, rstd, NPG0, 32, NT1 * 32);

  // ---- pool 1: keep K1 per graph ----
  topk_score<<<cdiv(NT1, TB), TB, 0, stream>>>(gat1, p1, score1, NT1, 32);
  fill_i32<<<1, 32, 0, stream>>>(ctrA, 0, BG);
  fill_i32<<<1, 32, 0, stream>>>(ctrB, 0, BG);
  radix_select<<<BG, TB, 0, stream>>>(score1, NPG0, KP1, Tenc, cntgt);
  topk_compact<<<cdiv(NT1, TB), TB, 0, stream>>>(gat1, score1, NPG0, KP1, 32, Tenc, cntgt,
                                                 ctrA, ctrB, newid1, x2, NT1);
  remap_edges<<<cdiv(NE, TB), TB, 0, stream>>>(src, dst, nullptr, newid1, src2, dst2, mask2, NE);

  // ================= Layer 2: GAT(32 -> 128, H=2, F=64) =================
  launch_gemm(0, x2, W2, nullptr, xp2, NT2, 128, 32, stream);
  gat_alpha<<<cdiv(NT2 * 2, TB), TB, 0, stream>>>(xp2, a_s2, alS2, NT2, 2, 64);
  gat_alpha<<<cdiv(NT2 * 2, TB), TB, 0, stream>>>(xp2, a_d2, alD2, NT2, 2, 64);
  fill_enc<<<cdiv(NT2 * 2, TB), TB, 0, stream>>>(emax2, -1e30f, NT2 * 2);
  fill_f32<<<cdiv(NT2 * 2, TB), TB, 0, stream>>>(den2, 0.f, NT2 * 2);
  fill_f32<<<cdiv(NT2 * 128, TB), TB, 0, stream>>>(gat2, 0.f, NT2 * 128);
  gat_edge_max<<<cdiv(NE + NT2, TB), TB, 0, stream>>>(src2, dst2, mask2, alS2, alD2, emax2,
                                                      NE, NT2, 2);
  gat_edge_msg<<<cdiv(NE + NT2, TB), TB, 0, stream>>>(src2, dst2, mask2, alS2, alD2, emax2,
                                                      den2, xp2, gat2, NE, NT2, 2, 64);
  gat_finalize<<<cdiv(NT2 * 128, TB), TB, 0, stream>>>(gat2, den2, b2, NT2, 2, 64);
  inorm_stats<<<BG * 128, TB, 0, stream>>>(gat2, KP1, 128, mu, rstd);
  inorm_apply_elu<<<cdiv(NT2 * 128, TB), TB, 0, stream>>>(gat2, mu, rstd, KP1, 128, NT2 * 128);

  // ---- pool 2: keep K2 per graph ----
  topk_score<<<cdiv(NT2, TB), TB, 0, stream>>>(gat2, p2, score2, NT2, 128);
  fill_i32<<<1, 32, 0, stream>>>(ctrA, 0, BG);
  fill_i32<<<1, 32, 0, stream>>>(ctrB, 0, BG);
  radix_select<<<BG, TB, 0, stream>>>(score2, KP1, KP2, Tenc, cntgt);
  topk_compact<<<cdiv(NT2, TB), TB, 0, stream>>>(gat2, score2, KP1, KP2, 128, Tenc, cntgt,
                                                 ctrA, ctrB, newid2, x3, NT2);
  remap_edges<<<cdiv(NE, TB), TB, 0, stream>>>(src2, dst2, mask2, newid2, src3, dst3, mask3, NE);

  // ================= Layer 3: GAT(128 -> 256, H=1, F=256) =================
  launch_gemm(0, x3, W3, nullptr, xp3, NT3, 256, 128, stream);
  gat_alpha<<<cdiv(NT3, TB), TB, 0, stream>>>(xp3, a_s3, alS3, NT3, 1, 256);
  gat_alpha<<<cdiv(NT3, TB), TB, 0, stream>>>(xp3, a_d3, alD3, NT3, 1, 256);
  fill_enc<<<cdiv(NT3, TB), TB, 0, stream>>>(emax3, -1e30f, NT3);
  fill_f32<<<cdiv(NT3, TB), TB, 0, stream>>>(den3, 0.f, NT3);
  fill_f32<<<cdiv(NT3 * 256, TB), TB, 0, stream>>>(gat3, 0.f, NT3 * 256);
  gat_edge_max<<<cdiv(NE + NT3, TB), TB, 0, stream>>>(src3, dst3, mask3, alS3, alD3, emax3,
                                                      NE, NT3, 1);
  gat_edge_msg<<<cdiv(NE + NT3, TB), TB, 0, stream>>>(src3, dst3, mask3, alS3, alD3, emax3,
                                                      den3, xp3, gat3, NE, NT3, 1, 256);
  gat_finalize<<<cdiv(NT3 * 256, TB), TB, 0, stream>>>(gat3, den3, b3, NT3, 1, 256);
  inorm_stats<<<BG * 256, TB, 0, stream>>>(gat3, KP2, 256, mu, rstd);
  inorm_apply_elu<<<cdiv(NT3 * 256, TB), TB, 0, stream>>>(gat3, mu, rstd, KP2, 256, NT3 * 256);

  // ================= Global attention readout + MLP =================
  launch_gemm(1, gat3, gw1, gb1, gate1, NT3, 64, 256, stream);                 // elu(x@gw1+gb1)
  dense_scalar<<<cdiv(NT3, TB), TB, 0, stream>>>(gate1, gw2, gb2, gateS, NT3, 1, 64, 0);
  launch_gemm(1, gat3, aw, ab, hbuf, NT3, 256, 256, stream);                   // elu(x@aw+ab)
  att_readout<<<BG, TB, 0, stream>>>(gateS, hbuf, gvec, KP2, 256);
  dense_scalar<<<cdiv(BG * 512, TB), TB, 0, stream>>>(gvec, w1, bb1, z1, BG, 512, 256, 1);
  dense_scalar<<<cdiv(BG * 128, TB), TB, 0, stream>>>(z1, w2, bb2, z2, BG, 128, 512, 1);
  dense_scalar<<<1, 128, 0, stream>>>(z2, w3, bb3, logits, BG, NCLS, 128, 0);
  log_softmax_rows<<<1, 32, 0, stream>>>(logits, (float*)d_out, BG, NCLS);
}
